// BiMambaEncoder_5978594476264
// MI455X (gfx1250) — matrix operand order repaired and checked by reading, not compiled, not run on hardware
//
#include <hip/hip_runtime.h>
#include <hip/hip_bf16.h>
#include <math.h>

#define BB 4
#define LL 576
#define DM 512
#define EDIM 1024
#define NLAYER 4
#define DSTATE 16
#define DTRANK 32
#define DCONV 4
#define MROWS (BB * LL) /* 2304 */
#define AST 136          /* padded LDS row stride in halves: 272B -> +4 banks/row */

typedef __attribute__((ext_vector_type(16))) _Float16 v16h;
typedef __attribute__((ext_vector_type(8)))  _Float16 v8h;
typedef __attribute__((ext_vector_type(8)))  float    v8f;
typedef __attribute__((ext_vector_type(4)))  int      v4i;

__device__ __forceinline__ float sigmoidf_(float x) { return 1.f / (1.f + __expf(-x)); }

// ---- CDNA5 async global->LDS (ASYNCcnt-tracked) ----
__device__ __forceinline__ void async_load_b128(const _Float16* g, _Float16* l) {
#if __has_builtin(__builtin_amdgcn_global_load_async_to_lds_b128)
  using gv4i = __attribute__((address_space(1))) v4i;
  using lv4i = __attribute__((address_space(3))) v4i;
  __builtin_amdgcn_global_load_async_to_lds_b128((gv4i*)g, (lv4i*)l, 0, 0);
#else
  unsigned lo = (unsigned)(size_t)(__attribute__((address_space(3))) void*)l;
  asm volatile("global_load_async_to_lds_b128 %0, %1, off" :: "v"(lo), "v"(g) : "memory");
#endif
}

template <int N>
__device__ __forceinline__ void wait_async() {
#if __has_builtin(__builtin_amdgcn_s_wait_asynccnt)
  __builtin_amdgcn_s_wait_asynccnt(N);
#else
  asm volatile("s_wait_asynccnt %0" :: "i"(N) : "memory");
#endif
}

// ---------------- elementwise: f32 -> f16 ----------------
__global__ void cvt_f16_kernel(const float* __restrict__ s, _Float16* __restrict__ d, int n) {
  int i = blockIdx.x * blockDim.x + threadIdx.x;
  if (i < n) d[i] = (_Float16)s[i];
}

// ---------------- feat (B,C,H,W) -> (B, L, C) f16 ----------------
__global__ void featT_kernel(const float* __restrict__ feat, _Float16* __restrict__ out) {
  const int C = 1024, H = 24, W = 24;
  int i = blockIdx.x * blockDim.x + threadIdx.x;
  if (i >= BB * LL * C) return;
  int c = i % C;
  int l = (i / C) % LL;
  int b = i / (C * LL);
  int h = l / W, w = l % W;
  out[i] = (_Float16)feat[(((size_t)b * C + c) * H + h) * W + w];
}

// ---------------- WMMA GEMM: C[M,N] = A[M,K] * W[N,K]^T (+epilogue) ----------------
// Block = 8 waves = 64(M) x 128(N) macro-tile; A strip staged in LDS via async loads,
// double buffered over 128-wide K chunks. Each wave: 1 N-tile x 4 M-tiles (W frag reuse).
// mode bit0: +bias[n]   bit1: softplus   bit2: +res[m,n]
__global__ __launch_bounds__(256) void wmma_gemm_kernel(
    const _Float16* __restrict__ A, const _Float16* __restrict__ Wt,
    const float* __restrict__ bias, const float* __restrict__ res,
    float* __restrict__ C, int M, int N, int K, int mode) {
  __shared__ _Float16 tileA[2][64 * AST];

  int t    = threadIdx.x;
  int lane = t & 31;
  int wv   = t >> 5;
  int ntn    = N >> 4;
  int strips = (ntn + 7) >> 3;
  int bm = blockIdx.x / strips;           // macro-tile row (64 rows)
  int sn = blockIdx.x % strips;           // macro-tile col strip (128 cols)
  int m0 = bm * 64;
  int tn = sn * 8 + wv;
  bool active = tn < ntn;

  int half = lane >> 4;                   // lane group 0..15 vs 16..31
  int l15  = lane & 15;
  int ka = half * 8;                      // A frag: K = {0..7,16..23} or {8..15,24..31}
  int kb = half * 8;                      // B frag: K = {0..7,16..23} or {8..15,24..31}
  long brow = (long)(tn * 16 + l15) * K;

  // stage one 64x128 K-chunk of A into LDS buffer b (4 async b128 per thread)
  auto stage = [&](int k0, int b) {
#pragma unroll
    for (int i = 0; i < 4; ++i) {
      int idx = t + 256 * i;              // 0..1023 segments (64 rows x 16 x 16B)
      int row = idx >> 4;
      int seg = idx & 15;
      int kk = k0 + seg * 8;
      if (kk + 8 > K) kk = K - 8;         // clamp (K % 8 == 0); stale data never read
      async_load_b128(A + (size_t)(m0 + row) * K + kk, &tileA[b][row * AST + seg * 8]);
    }
  };

  v8f acc[4] = {{}, {}, {}, {}};
  int nch = (K + 127) >> 7;
  stage(0, 0);
  for (int c = 0; c < nch; ++c) {
    int k0 = c << 7;
    int buf = c & 1;
    if (c + 1 < nch) { stage(k0 + 128, buf ^ 1); wait_async<4>(); }
    else             { wait_async<0>(); }
    __syncthreads();
    if (active) {
      // one 32-wide K step: batch all LDS/global frag loads, then 4 independent WMMAs
      auto step = [&](int j) {
        v8h b0 = *(const v8h*)(Wt + brow + k0 + 32 * j + kb);
        v8h b1 = *(const v8h*)(Wt + brow + k0 + 32 * j + kb + 16);
        v8h a0[4], a1[4];
#pragma unroll
        for (int mt = 0; mt < 4; ++mt) {
          const _Float16* As = &tileA[buf][(mt * 16 + l15) * AST + 32 * j + ka];
          a0[mt] = *(const v8h*)As;
          a1[mt] = *(const v8h*)(As + 16);
        }
        v16h bf;
#pragma unroll
        for (int i = 0; i < 8; ++i) { bf[i] = b0[i]; bf[i + 8] = b1[i]; }
#pragma unroll
        for (int mt = 0; mt < 4; ++mt) {
          v16h af;
#pragma unroll
          for (int i = 0; i < 8; ++i) { af[i] = a0[mt][i]; af[i + 8] = a1[mt][i]; }
          acc[mt] = __builtin_amdgcn_wmma_f32_16x16x32_f16(false, af, false, bf, (short)0,
                                                           acc[mt], false, false);
        }
      };
      int rem = K - k0;
      if (rem >= 128) {                   // straight-line fast path (all full chunks)
        step(0); step(1); step(2); step(3);
      } else {                            // tail chunk (only the K=32 dt GEMM hits this)
        for (int j = 0; j * 32 < rem; ++j) step(j);
      }
    }
    __syncthreads();
  }

  if (!active) return;
  int nn = tn * 16 + l15;
  float bv = (mode & 1) ? bias[nn] : 0.f;
#pragma unroll
  for (int mt = 0; mt < 4; ++mt) {
#pragma unroll
    for (int r = 0; r < 8; ++r) {
      int mm = m0 + mt * 16 + r + half * 8;   // C layout: lanes16-31 hold M=r+8
      float v = acc[mt][r] + bv;
      if (mode & 2) v = (v > 20.f) ? v : log1pf(__expf(v));
      long o = (long)mm * N + nn;
      if (mode & 4) v += res[o];
      C[o] = v;
    }
  }
}

// ---------------- +pos_emb, produce forward copy and L-flipped copy ----------------
__global__ void posflip_kernel(const float* __restrict__ x0, const float* __restrict__ pos,
                               float* __restrict__ xf, float* __restrict__ xb) {
  int i = blockIdx.x * blockDim.x + threadIdx.x;
  if (i >= MROWS * DM) return;
  int n = i % DM;
  int l = (i / DM) % LL;
  int b = i / (DM * LL);
  float v = x0[i] + pos[l * DM + n];
  xf[i] = v;
  xb[((size_t)(b * LL + (LL - 1 - l))) * DM + n] = v;
}

// ---------------- rmsnorm row (512) -> f16 ----------------
__global__ __launch_bounds__(256) void rmsnorm_kernel(const float* __restrict__ x,
                                                      const float* __restrict__ w,
                                                      _Float16* __restrict__ out) {
  __shared__ float red[256];
  int row = blockIdx.x, t = threadIdx.x;
  const float* xr = x + (size_t)row * DM;
  float v0 = xr[t], v1 = xr[t + 256];
  red[t] = v0 * v0 + v1 * v1;
  __syncthreads();
  for (int s = 128; s > 0; s >>= 1) { if (t < s) red[t] += red[t + s]; __syncthreads(); }
  float scale = rsqrtf(red[0] / DM + 1e-5f);
  out[(size_t)row * DM + t]       = (_Float16)(v0 * scale * w[t]);
  out[(size_t)row * DM + t + 256] = (_Float16)(v1 * scale * w[t + 256]);
}

// ---------------- depthwise causal conv (D_CONV=4) + bias + SiLU ----------------
__global__ void conv_silu_kernel(const float* __restrict__ xz, const float* __restrict__ cw,
                                 const float* __restrict__ cb, float* __restrict__ xc,
                                 _Float16* __restrict__ xch) {
  int i = blockIdx.x * blockDim.x + threadIdx.x;
  if (i >= MROWS * EDIM) return;
  int e = i % EDIM;
  int m = i / EDIM;
  int l = m % LL;
  float acc = cb[e];
#pragma unroll
  for (int j = 0; j < DCONV; ++j) {
    int lj = l - (DCONV - 1) + j;
    if (lj >= 0) acc += xz[(size_t)(m - (DCONV - 1) + j) * (2 * EDIM) + e] * cw[e * DCONV + j];
  }
  acc = acc * sigmoidf_(acc);
  xc[i]  = acc;
  xch[i] = (_Float16)acc;
}

// ---------------- dbc[:, :32] -> f16 for dt GEMM ----------------
__global__ void slice_dt_kernel(const float* __restrict__ dbc, _Float16* __restrict__ out) {
  int i = blockIdx.x * blockDim.x + threadIdx.x;
  if (i >= MROWS * DTRANK) return;
  int r = i % DTRANK, m = i / DTRANK;
  out[i] = (_Float16)dbc[(size_t)m * (DTRANK + 2 * DSTATE) + r];
}

// ---------------- selective scan: one thread per (b,e) channel ----------------
__global__ __launch_bounds__(256) void scan_kernel(const float* __restrict__ xc,
                                                   const float* __restrict__ delta,
                                                   const float* __restrict__ dbc,
                                                   const float* __restrict__ Alog,
                                                   const float* __restrict__ Dp,
                                                   float* __restrict__ y) {
  int tid = blockIdx.x * blockDim.x + threadIdx.x;
  if (tid >= BB * EDIM) return;
  int e = tid % EDIM, b = tid / EDIM;
  float a[DSTATE], h[DSTATE];
#pragma unroll
  for (int n = 0; n < DSTATE; ++n) { a[n] = -__expf(Alog[e * DSTATE + n]); h[n] = 0.f; }
  float Dd = Dp[e];
  for (int l = 0; l < LL; ++l) {
    size_t row = (size_t)b * LL + l;
    float d  = delta[row * EDIM + e];
    float xv = xc[row * EDIM + e];
    const float* bc = dbc + row * 64;
    float dx = d * xv;
    float yv = 0.f;
#pragma unroll
    for (int n = 0; n < DSTATE; ++n) {
      h[n] = __expf(d * a[n]) * h[n] + dx * bc[DTRANK + n];
      yv  += h[n] * bc[DTRANK + DSTATE + n];
    }
    y[row * EDIM + e] = yv + Dd * xv;
  }
}

// ---------------- y_scan * silu(z) -> f16 ----------------
__global__ void gate_kernel(const float* __restrict__ ysc, const float* __restrict__ xz,
                            _Float16* __restrict__ yg) {
  int i = blockIdx.x * blockDim.x + threadIdx.x;
  if (i >= MROWS * EDIM) return;
  int e = i % EDIM, m = i / EDIM;
  float z = xz[(size_t)m * (2 * EDIM) + EDIM + e];
  yg[i] = (_Float16)(ysc[i] * z * sigmoidf_(z));
}

// ---------------- layernorm(y_f + rev(y_b)) then mean over L via atomics ----------------
__global__ __launch_bounds__(256) void final_kernel(const float* __restrict__ xf,
                                                    const float* __restrict__ xb,
                                                    const float* __restrict__ g,
                                                    const float* __restrict__ bta,
                                                    float* __restrict__ out) {
  __shared__ float red[256];
  int m = blockIdx.x, t = threadIdx.x;
  int l = m % LL, b = m / LL;
  size_t rf = (size_t)m * DM;
  size_t rb = ((size_t)b * LL + (LL - 1 - l)) * DM;
  float v0 = xf[rf + t] + xb[rb + t];
  float v1 = xf[rf + t + 256] + xb[rb + t + 256];
  red[t] = v0 + v1;
  __syncthreads();
  for (int s = 128; s > 0; s >>= 1) { if (t < s) red[t] += red[t + s]; __syncthreads(); }
  float mean = red[0] / DM;
  __syncthreads();
  float d0 = v0 - mean, d1 = v1 - mean;
  red[t] = d0 * d0 + d1 * d1;
  __syncthreads();
  for (int s = 128; s > 0; s >>= 1) { if (t < s) red[t] += red[t + s]; __syncthreads(); }
  float inv = rsqrtf(red[0] / DM + 1e-5f);
  atomicAdd(&out[b * DM + t],       (d0 * inv * g[t]       + bta[t])       * (1.f / LL));
  atomicAdd(&out[b * DM + t + 256], (d1 * inv * g[t + 256] + bta[t + 256]) * (1.f / LL));
}

// =====================================================================
extern "C" void kernel_launch(void* const* d_in, const int* in_sizes, int n_in,
                              void* d_out, int out_size, void* d_ws, size_t ws_size,
                              hipStream_t stream) {
  (void)in_sizes; (void)n_in; (void)ws_size;
  const float* feat  = (const float*)d_in[0];
  const float* projw = (const float*)d_in[1];
  const float* projb = (const float*)d_in[2];
  const float* pos   = (const float*)d_in[3];
  struct Params {
    const float *in_w, *conv_w, *conv_b, *xproj_w, *dt_w, *dt_b, *A_log, *D, *out_w, *norm_w;
  } P[2];
  for (int d = 0; d < 2; ++d) {
    int base = 4 + d * 10;
    P[d].in_w    = (const float*)d_in[base + 0];
    P[d].conv_w  = (const float*)d_in[base + 1];
    P[d].conv_b  = (const float*)d_in[base + 2];
    P[d].xproj_w = (const float*)d_in[base + 3];
    P[d].dt_w    = (const float*)d_in[base + 4];
    P[d].dt_b    = (const float*)d_in[base + 5];
    P[d].A_log   = (const float*)d_in[base + 6];
    P[d].D       = (const float*)d_in[base + 7];
    P[d].out_w   = (const float*)d_in[base + 8];
    P[d].norm_w  = (const float*)d_in[base + 9];
  }
  const float* lng = (const float*)d_in[24];
  const float* lnb = (const float*)d_in[25];

  char* ws = (char*)d_ws;
  size_t off = 0;
  auto alloc = [&](size_t bytes) -> void* {
    off = (off + 255) & ~(size_t)255;
    void* p = ws + off;
    off += bytes;
    return p;
  };
  float* x0  = (float*)alloc((size_t)MROWS * DM * 4);
  float* xF  = (float*)alloc((size_t)MROWS * DM * 4);
  float* xB  = (float*)alloc((size_t)MROWS * DM * 4);
  float* xz  = (float*)alloc((size_t)MROWS * 2 * EDIM * 4);
  float* xc  = (float*)alloc((size_t)MROWS * EDIM * 4);
  float* dbc = (float*)alloc((size_t)MROWS * 64 * 4);
  float* dlt = (float*)alloc((size_t)MROWS * EDIM * 4);
  float* ysc = (float*)alloc((size_t)MROWS * EDIM * 4);
  _Float16* featH  = (_Float16*)alloc((size_t)MROWS * 1024 * 2);
  _Float16* xnH    = (_Float16*)alloc((size_t)MROWS * DM * 2);
  _Float16* xcH    = (_Float16*)alloc((size_t)MROWS * EDIM * 2);
  _Float16* dtH    = (_Float16*)alloc((size_t)MROWS * DTRANK * 2);
  _Float16* ygH    = (_Float16*)alloc((size_t)MROWS * EDIM * 2);
  _Float16* projwH = (_Float16*)alloc((size_t)DM * 1024 * 2);
  _Float16 *inwH[2], *xpH[2], *dtwH[2], *owH[2];
  for (int d = 0; d < 2; ++d) {
    inwH[d] = (_Float16*)alloc((size_t)NLAYER * 2 * EDIM * DM * 2);
    xpH[d]  = (_Float16*)alloc((size_t)NLAYER * 64 * EDIM * 2);
    dtwH[d] = (_Float16*)alloc((size_t)NLAYER * EDIM * DTRANK * 2);
    owH[d]  = (_Float16*)alloc((size_t)NLAYER * DM * EDIM * 2);
  }

  auto cvt = [&](const float* s, _Float16* dd, size_t n) {
    cvt_f16_kernel<<<(n + 255) / 256, 256, 0, stream>>>(s, dd, (int)n);
  };
  cvt(projw, projwH, (size_t)DM * 1024);
  for (int d = 0; d < 2; ++d) {
    cvt(P[d].in_w,    inwH[d], (size_t)NLAYER * 2 * EDIM * DM);
    cvt(P[d].xproj_w, xpH[d],  (size_t)NLAYER * 64 * EDIM);
    cvt(P[d].dt_w,    dtwH[d], (size_t)NLAYER * EDIM * DTRANK);
    cvt(P[d].out_w,   owH[d],  (size_t)NLAYER * DM * EDIM);
  }
  featT_kernel<<<(MROWS * 1024 + 255) / 256, 256, 0, stream>>>(feat, featH);

  auto gemm = [&](const _Float16* A, const _Float16* Wt, const float* bias, const float* res,
                  float* C, int M, int N, int K, int mode) {
    int strips = ((N / 16) + 7) / 8;
    int blocks = (M / 64) * strips;
    wmma_gemm_kernel<<<blocks, 256, 0, stream>>>(A, Wt, bias, res, C, M, N, K, mode);
  };

  // input projection + pos-emb, make forward & flipped residual streams
  gemm(featH, projwH, projb, nullptr, x0, MROWS, DM, 1024, 1);
  posflip_kernel<<<(MROWS * DM + 255) / 256, 256, 0, stream>>>(x0, pos, xF, xB);

  for (int d = 0; d < 2; ++d) {
    float* x = d ? xB : xF;
    for (int i = 0; i < NLAYER; ++i) {
      const _Float16* inw = inwH[d] + (size_t)i * 2 * EDIM * DM;
      const _Float16* xpw = xpH[d]  + (size_t)i * 64 * EDIM;
      const _Float16* dtw = dtwH[d] + (size_t)i * EDIM * DTRANK;
      const _Float16* ow  = owH[d]  + (size_t)i * DM * EDIM;
      const float* cw  = P[d].conv_w + (size_t)i * EDIM * DCONV;
      const float* cb  = P[d].conv_b + (size_t)i * EDIM;
      const float* dtb = P[d].dt_b   + (size_t)i * EDIM;
      const float* Al  = P[d].A_log  + (size_t)i * EDIM * DSTATE;
      const float* Dp  = P[d].D      + (size_t)i * EDIM;
      const float* nw  = P[d].norm_w + (size_t)i * DM;

      rmsnorm_kernel<<<MROWS, 256, 0, stream>>>(x, nw, xnH);
      gemm(xnH, inw, nullptr, nullptr, xz, MROWS, 2 * EDIM, DM, 0);
      conv_silu_kernel<<<(MROWS * EDIM + 255) / 256, 256, 0, stream>>>(xz, cw, cb, xc, xcH);
      gemm(xcH, xpw, nullptr, nullptr, dbc, MROWS, 64, EDIM, 0);
      slice_dt_kernel<<<(MROWS * DTRANK + 255) / 256, 256, 0, stream>>>(dbc, dtH);
      gemm(dtH, dtw, dtb, nullptr, dlt, MROWS, EDIM, DTRANK, 1 | 2);
      scan_kernel<<<(BB * EDIM + 255) / 256, 256, 0, stream>>>(xc, dlt, dbc, Al, Dp, ysc);
      gate_kernel<<<(MROWS * EDIM + 255) / 256, 256, 0, stream>>>(ysc, xz, ygH);
      gemm(ygH, ow, nullptr, x, x, MROWS, DM, EDIM, 4); // residual add in-place
    }
  }

  (void)hipMemsetAsync(d_out, 0, (size_t)out_size * sizeof(float), stream);
  final_kernel<<<MROWS, 256, 0, stream>>>(xF, xB, lng, lnb, (float*)d_out);
}
